// GINwithTopK_21277267984632
// MI455X (gfx1250) — matrix operand order repaired and checked by reading, not compiled, not run on hardware
//
#include <hip/hip_runtime.h>
#include <hip/hip_bf16.h>
#include <math.h>

// ---------------- problem constants (match reference) ----------------
#define GRAPHS 256
#define NPG    256                  // nodes per graph
#define TOTN   (GRAPHS*NPG)         // 65536
#define DEG    16
#define EPG    (NPG*DEG)            // 4096 edges per graph (contiguous block)
#define NEDGE  (TOTN*DEG)           // 1048576
#define HD     128
#define NLAY   5
#define K1P    205
#define K2P    164

typedef __bf16 bf16;
typedef __attribute__((ext_vector_type(16))) __bf16 v16bf;
typedef __attribute__((ext_vector_type(8)))  __bf16 v8bf;
typedef __attribute__((ext_vector_type(8)))  float  v8f;

__device__ __host__ constexpr float BN_RSQ = 0.99999500003749975f; // 1/sqrt(1+1e-5)

// ---------------- dynamic LDS layout (bytes) ----------------
#define OFF_ACCF   0                          // float[256*128]  h accumulator / GEMM1 out
#define OFF_ESRC   (OFF_ACCF + NPG*HD*4)      // int[4096]
#define OFF_EDST   (OFF_ESRC + EPG*4)         // int[4096]
#define OFF_SMASK  (OFF_EDST + EPG*4)         // float[256]
#define OFF_PSUM   (OFF_SMASK + NPG*4)        // float[128] pooling column sums
#define OFF_SCAL   (OFF_PSUM + HD*4)          // float[8] (padded to keep 32B alignment)
#define OFF_ABUF   (OFF_SCAL + 32)            // bf16[256*128]  WMMA A operand (32B aligned)
#define OFF_WBUF   (OFF_ABUF + NPG*HD*2)      // bf16[128*128]  W staged fragment-major
#define LDS_TOTAL  (OFF_WBUF + HD*HD*2)       // ~258 KB < 320 KB per workgroup

// ---------------- WMMA fragment loaders (CDNA5 16-bit layouts) ----------------
// A 16x32 bf16 (row-major LDS): lanes 0-15 row M=lane, elems 0..7 -> K 0..7,
// elems 8..15 -> K 16..23; lanes 16-31 same rows, K ranges +8.
// Two contiguous 16B chunks per lane -> 2x ds_load_b128.
__device__ __forceinline__ v16bf frag_a(const bf16* A, int tm, int kc, int lane) {
  const bf16* p = A + (tm*16 + (lane & 15))*HD + kc*32 + ((lane >> 4) << 3);
  v8bf lo = *(const v8bf*)p;
  v8bf hi = *(const v8bf*)(p + 16);
  return __builtin_shufflevector(lo, hi, 0,1,2,3,4,5,6,7,8,9,10,11,12,13,14,15);
}
// B operand staged FRAGMENT-MAJOR in LDS: for (kc,tn,lane) the 16 values a lane
// feeds to the WMMA are contiguous (32B) -> one v16bf load (2x ds_load_b128).
__device__ __forceinline__ v16bf frag_b(const bf16* Wf, int tn, int kc, int lane) {
  return *(const v16bf*)(Wf + ((((kc << 3) + tn) << 5) + lane)*16);
}

// Stage W (global f32, K-major [K][N]) into LDS fragment-major bf16.
// Hardware B layout: lanes 0-15 -> col N=tn*16+lane, K = kc*32 + (i&7) + 16*(i>>3);
//                    lanes 16-31 -> col N=tn*16+lane-16, K = kc*32 + 8 + (i&7) + 16*(i>>3).
__device__ __forceinline__ void stage_w(bf16* wf, const float* __restrict__ W, int tid) {
  for (int idx = tid; idx < HD*HD; idx += 256) {
    int i    = idx & 15;
    int lane = (idx >> 4) & 31;
    int tn   = (idx >> 9) & 7;
    int kc   = idx >> 12;
    int k    = kc*32 + ((lane >> 4) << 3) + (i & 7) + ((i >> 3) << 4);
    int c    = tn*16 + (lane & 15);
    wf[idx] = (bf16)W[k*HD + c];
  }
}

__global__ void init_mask_kernel(float* __restrict__ m) {
  m[blockIdx.x * blockDim.x + threadIdx.x] = 1.0f;
}

// ---------------- fused GIN layer: one workgroup per graph ----------------
__global__ __launch_bounds__(256)
void gin_layer_kernel(const float* __restrict__ x_in,
                      float* __restrict__ x_out,
                      const float* __restrict__ mask,
                      const int* __restrict__ esrc,
                      const int* __restrict__ edst,
                      const float* __restrict__ W1, const float* __restrict__ b1,
                      const float* __restrict__ W2, const float* __restrict__ b2,
                      const float* __restrict__ bn_g, const float* __restrict__ bn_b,
                      float* __restrict__ xs, int layer)
{
  extern __shared__ char lds_raw[];
  float* accf  = (float*)(lds_raw + OFF_ACCF);
  int*   sS    = (int*)  (lds_raw + OFF_ESRC);
  int*   sD    = (int*)  (lds_raw + OFF_EDST);
  float* smask = (float*)(lds_raw + OFF_SMASK);
  float* psum  = (float*)(lds_raw + OFF_PSUM);
  float* scal  = (float*)(lds_raw + OFF_SCAL);
  bf16*  abuf  = (bf16*) (lds_raw + OFF_ABUF);
  bf16*  wbuf  = (bf16*) (lds_raw + OFF_WBUF);

  const int tid  = threadIdx.x;
  const int lane = tid & 31;
  const int wv   = tid >> 5;     // 8 waves; wave w owns output column block [16w,16w+16)
  const int g    = blockIdx.x;
  const int base = g * NPG;

  smask[tid] = mask[base + tid];
  { // stage edge list (graph-local indices)
    const int* s0 = esrc + (size_t)g*EPG;
    const int* d0 = edst + (size_t)g*EPG;
    for (int i = tid; i < EPG; i += 256) { sS[i] = s0[i] - base; sD[i] = d0[i] - base; }
  }
  stage_w(wbuf, W1, tid);                                        // W1 fragment-major
  __syncthreads();
  if (tid == 0) { float c = 0.f; for (int i = 0; i < NPG; ++i) c += smask[i]; scal[0] = c; }
  // stage x: accf = x (GIN: h = x + agg), abuf = bf16(x * mask) messages
  for (int idx = tid; idx < NPG*HD; idx += 256) {
    int n = idx >> 7;
    float v = x_in[(size_t)base*HD + idx];
    accf[idx] = v;
    abuf[idx] = (bf16)(v * smask[n]);
  }
  __syncthreads();
  // scatter-sum over edges: lanes map to consecutive features -> conflict-free banks
  {
    const int f = tid & (HD - 1);
    for (int e = (tid >> 7); e < EPG; e += 2) {
      float mv = (float)abuf[sS[e]*HD + f];
      atomicAdd(&accf[sD[e]*HD + f], mv);    // ds_add_f32
    }
  }
  __syncthreads();
  for (int idx = tid; idx < NPG*HD; idx += 256) abuf[idx] = (bf16)accf[idx]; // A = h
  __syncthreads();

  bf16* hbuf = (bf16*)accf;   // reuse f32 region for GEMM1 bf16 output (16B aligned)
  const int tn  = wv;
  const int col = tn*16 + (lane & 15);
  const int rsh = (lane >> 4) << 3;

  // GEMM1: hbuf = relu(h @ W1 + b1), bf16  (16 M-tiles per wave, K=128 in 4 WMMA steps)
  {
    const float bias = b1[col];
    for (int tm = 0; tm < 16; ++tm) {
      v8f c = {};
#pragma unroll
      for (int kc = 0; kc < 4; ++kc) {
        v16bf af = frag_a(abuf, tm, kc, lane);
        v16bf bfr = frag_b(wbuf, tn, kc, lane);
        c = __builtin_amdgcn_wmma_f32_16x16x32_bf16(false, af, false, bfr,
                                                    (short)0, c, false, false);
      }
#pragma unroll
      for (int v = 0; v < 8; ++v) {
        int row = tm*16 + v + rsh;
        hbuf[row*HD + col] = (bf16)fmaxf(c[v] + bias, 0.0f);
      }
    }
  }
  __syncthreads();
  stage_w(wbuf, W2, tid);                                        // W2 fragment-major
  if (tid < HD) psum[tid] = 0.0f;
  __syncthreads();

  // GEMM2 + bias + relu + BN + outer relu + mask; store x_out and mean-pool sums
  {
    const float bias = b2[col];
    const float scl  = bn_g[col] * BN_RSQ;
    const float sft  = bn_b[col];
    float csum = 0.0f;
    for (int tm = 0; tm < 16; ++tm) {
      v8f c = {};
#pragma unroll
      for (int kc = 0; kc < 4; ++kc) {
        v16bf af = frag_a(hbuf, tm, kc, lane);
        v16bf bfr = frag_b(wbuf, tn, kc, lane);
        c = __builtin_amdgcn_wmma_f32_16x16x32_bf16(false, af, false, bfr,
                                                    (short)0, c, false, false);
      }
#pragma unroll
      for (int v = 0; v < 8; ++v) {
        int row = tm*16 + v + rsh;
        float val = fmaxf(c[v] + bias, 0.0f);
        val = fmaxf(val*scl + sft, 0.0f) * smask[row];
        x_out[(size_t)(base + row)*HD + col] = val;
        csum += val;
      }
    }
    atomicAdd(&psum[col], csum);   // wave-constant col; 2 lanes per column
  }
  __syncthreads();
  if (tid < HD) xs[(size_t)g*(NLAY*HD) + layer*HD + tid] = psum[tid] / scal[0];
}

// ---------------- TopK pooling: per-graph bitonic sort of 256 keys ----------------
__global__ __launch_bounds__(256)
void topk_kernel(float* __restrict__ x, float* __restrict__ mask,
                 const float* __restrict__ w, int k)
{
  __shared__ float sw[HD];
  __shared__ unsigned long long keys[NPG];
  __shared__ float sraw[NPG];
  __shared__ int ssel[NPG];
  __shared__ float snorm;
  const int tid  = threadIdx.x;
  const int g    = blockIdx.x;
  const int base = g * NPG;

  if (tid < HD) sw[tid] = w[tid];
  ssel[tid] = 0;
  __syncthreads();
  if (tid == 0) { float s = 0.f; for (int i = 0; i < HD; ++i) s += sw[i]*sw[i]; snorm = sqrtf(s); }
  __syncthreads();

  float r = 0.f;
  {
    const float* xr = x + (size_t)(base + tid)*HD;
    for (int f = 0; f < HD; ++f) r += xr[f]*sw[f];
  }
  r /= snorm;
  sraw[tid] = r;
  float sc = (mask[base + tid] > 0.f) ? r : -1e9f;
  unsigned u = __float_as_uint(sc);
  u = (u & 0x80000000u) ? ~u : (u | 0x80000000u);          // monotone float->uint
  keys[tid] = ((unsigned long long)u << 32) | (unsigned)(NPG - 1 - tid); // tie: low idx wins
  __syncthreads();

  for (int size = 2; size <= NPG; size <<= 1)
    for (int stride = size >> 1; stride > 0; stride >>= 1) {
      int i = tid, j = i ^ stride;
      if (j > i) {
        unsigned long long a = keys[i], b = keys[j];
        bool asc = ((i & size) == 0);
        if ((a > b) == asc) { keys[i] = b; keys[j] = a; }
      }
      __syncthreads();
    }

  if (tid >= NPG - k) {
    int node = (NPG - 1) - (int)(keys[tid] & 0xFFFFFFFFull);
    ssel[node] = 1;
  }
  __syncthreads();
  float m = ssel[tid] ? 1.0f : 0.0f;
  mask[base + tid] = m;
  float fac = tanhf(sraw[tid]) * m;
  float* xw = x + (size_t)(base + tid)*HD;
  for (int f = 0; f < HD; ++f) xw[f] *= fac;
}

// ---------------- MLP head + log_softmax: one block per graph ----------------
__global__ __launch_bounds__(256)
void mlp_head_kernel(const float* __restrict__ xs,
                     const float* __restrict__ lin1W, const float* __restrict__ lin1b,
                     const float* __restrict__ fc1W,  const float* __restrict__ fc1b,
                     const float* __restrict__ bn1g,  const float* __restrict__ bn1b,
                     const float* __restrict__ fc2W,  const float* __restrict__ fc2b,
                     const float* __restrict__ bn2g,  const float* __restrict__ bn2b,
                     const float* __restrict__ lin2W, const float* __restrict__ lin2b,
                     float* __restrict__ out)
{
  __shared__ float sx[NLAY*HD];
  __shared__ float h1[HD];
  __shared__ float h2[32];
  __shared__ float h3[512];
  __shared__ float o[16];
  const int tid = threadIdx.x;
  const int g   = blockIdx.x;
  for (int i = tid; i < NLAY*HD; i += 256) sx[i] = xs[(size_t)g*NLAY*HD + i];
  __syncthreads();
  if (tid < HD) {
    float s = lin1b[tid];
    for (int k = 0; k < NLAY*HD; ++k) s += sx[k]*lin1W[k*HD + tid];
    h1[tid] = fmaxf(s, 0.f);
  }
  __syncthreads();
  if (tid < 32) {
    float s = fc1b[tid];
    for (int k = 0; k < HD; ++k) s += h1[k]*fc1W[k*32 + tid];
    s = fmaxf(s, 0.f);
    h2[tid] = s*(bn1g[tid]*BN_RSQ) + bn1b[tid];
  }
  __syncthreads();
  for (int j = tid; j < 512; j += 256) {
    float s = fc2b[j];
    for (int k = 0; k < 32; ++k) s += h2[k]*fc2W[k*512 + j];
    s = fmaxf(s, 0.f);
    h3[j] = s*(bn2g[j]*BN_RSQ) + bn2b[j];
  }
  __syncthreads();
  if (tid < 10) {
    float s = lin2b[tid];
    for (int k = 0; k < 512; ++k) s += h3[k]*lin2W[k*10 + tid];
    o[tid] = s;
  }
  __syncthreads();
  if (tid == 0) {
    float mx = o[0];
    for (int c = 1; c < 10; ++c) mx = fmaxf(mx, o[c]);
    float se = 0.f;
    for (int c = 0; c < 10; ++c) se += expf(o[c] - mx);
    float l = logf(se);
    for (int c = 0; c < 10; ++c) out[g*10 + c] = o[c] - mx - l;
  }
}

// ---------------- host orchestration ----------------
extern "C" void kernel_launch(void* const* d_in, const int* in_sizes, int n_in,
                              void* d_out, int out_size, void* d_ws, size_t ws_size,
                              hipStream_t stream)
{
  (void)in_sizes; (void)n_in; (void)out_size; (void)ws_size;
  // Input order (setup_inputs dict insertion order, leaves flattened):
  // 0:x  1:edge_index  2..31: convs[i]{W1,b1,W2,b2,g,b}  32,33: pools
  // 34..45: lin1W,lin1b, fc1W,fc1b, bn1g,bn1b, fc2W,fc2b, bn2g,bn2b, lin2W,lin2b
  const float* x   = (const float*)d_in[0];
  const int*   ei  = (const int*)  d_in[1];
  const int*   src = ei;
  const int*   dst = ei + NEDGE;

  const float* cW1[NLAY]; const float* cb1[NLAY]; const float* cW2[NLAY];
  const float* cb2[NLAY]; const float* cgm[NLAY]; const float* cbt[NLAY];
  for (int i = 0; i < NLAY; ++i) {
    cW1[i] = (const float*)d_in[2 + 6*i + 0];
    cb1[i] = (const float*)d_in[2 + 6*i + 1];
    cW2[i] = (const float*)d_in[2 + 6*i + 2];
    cb2[i] = (const float*)d_in[2 + 6*i + 3];
    cgm[i] = (const float*)d_in[2 + 6*i + 4];
    cbt[i] = (const float*)d_in[2 + 6*i + 5];
  }
  const float* pool0 = (const float*)d_in[32];
  const float* pool1 = (const float*)d_in[33];
  const float* lin1W = (const float*)d_in[34];
  const float* lin1b = (const float*)d_in[35];
  const float* fc1W  = (const float*)d_in[36];
  const float* fc1b  = (const float*)d_in[37];
  const float* bn1g  = (const float*)d_in[38];
  const float* bn1b  = (const float*)d_in[39];
  const float* fc2W  = (const float*)d_in[40];
  const float* fc2b  = (const float*)d_in[41];
  const float* bn2g  = (const float*)d_in[42];
  const float* bn2b  = (const float*)d_in[43];
  const float* lin2W = (const float*)d_in[44];
  const float* lin2b = (const float*)d_in[45];

  float* ws  = (float*)d_ws;
  float* xA  = ws;                          // [TOTN,HD]
  float* xB  = xA + (size_t)TOTN*HD;        // [TOTN,HD]
  float* mb  = xB + (size_t)TOTN*HD;        // [TOTN] mask
  float* xsb = mb + TOTN;                   // [GRAPHS, NLAY*HD] pooled features

  const size_t lds = LDS_TOTAL;

  init_mask_kernel<<<TOTN/256, 256, 0, stream>>>(mb);
  // layer 0 .. 4 with TopK after layers 1 and 3 (matches reference schedule)
  gin_layer_kernel<<<GRAPHS, 256, lds, stream>>>(x,  xA, mb, src, dst,
      cW1[0], cb1[0], cW2[0], cb2[0], cgm[0], cbt[0], xsb, 0);
  gin_layer_kernel<<<GRAPHS, 256, lds, stream>>>(xA, xB, mb, src, dst,
      cW1[1], cb1[1], cW2[1], cb2[1], cgm[1], cbt[1], xsb, 1);
  topk_kernel<<<GRAPHS, 256, 0, stream>>>(xB, mb, pool0, K1P);
  gin_layer_kernel<<<GRAPHS, 256, lds, stream>>>(xB, xA, mb, src, dst,
      cW1[2], cb1[2], cW2[2], cb2[2], cgm[2], cbt[2], xsb, 2);
  gin_layer_kernel<<<GRAPHS, 256, lds, stream>>>(xA, xB, mb, src, dst,
      cW1[3], cb1[3], cW2[3], cb2[3], cgm[3], cbt[3], xsb, 3);
  topk_kernel<<<GRAPHS, 256, 0, stream>>>(xB, mb, pool1, K2P);
  gin_layer_kernel<<<GRAPHS, 256, lds, stream>>>(xB, xA, mb, src, dst,
      cW1[4], cb1[4], cW2[4], cb2[4], cgm[4], cbt[4], xsb, 4);

  mlp_head_kernel<<<GRAPHS, 256, 0, stream>>>(xsb, lin1W, lin1b, fc1W, fc1b,
      bn1g, bn1b, fc2W, fc2b, bn2g, bn2b, lin2W, lin2b, (float*)d_out);
}
